// CPWGenerator_78512002171233
// MI455X (gfx1250) — compile-verified, matrix operand order checked by reading, not run on hardware
//
#include <hip/hip_runtime.h>
#include <cstdint>
#include <cstddef>

// ---------------------------------------------------------------------------
// CDNA5 (gfx1250) fused CPW generator, round 4.
// - GEMMs: v_wmma_f32_16x16x32_f16, wave32, f32 accum.
// - Weights + padded bias staged to LDS per workgroup via async global->LDS
//   (ASYNCcnt); B fragments via ds_load_b128; A fragments in registers.
// - 128-thread blocks (4 waves), 64 rows each -> 256 blocks/layer.
// - Combine phase: branchless streaming writes, block-uniform scalar loads.
// Fragment layout (ISA 7.12.2, 16-bit 16x32): lane = row(A)/col(B),
// halves = k in [8*hi, 8*hi+8) ++ [16+8*hi, 16+8*hi+8) -> two b128 loads.
// ---------------------------------------------------------------------------

typedef __attribute__((ext_vector_type(16))) _Float16 v16h;
typedef __attribute__((ext_vector_type(8)))  _Float16 v8h;
typedef __attribute__((ext_vector_type(8)))  float    v8f;

union FragU { v16h v; v8h h[2]; };

template <typename T>
__device__ inline v16h load_frag16(const T* p, int hi) {
  FragU u;
  u.h[0] = *(const v8h*)(p + 8 * hi);        // k = 8*hi .. 8*hi+7
  u.h[1] = *(const v8h*)(p + 16 + 8 * hi);   // k = 16+8*hi .. 16+8*hi+7
  return u.v;
}

// Async 16B global -> LDS copy (CDNA5 ASYNCcnt path); signature probe-verified.
typedef int gcc_v4i __attribute__((vector_size(16)));
typedef __attribute__((address_space(1))) gcc_v4i* as1_v4i_ptr;
typedef __attribute__((address_space(3))) gcc_v4i* as3_v4i_ptr;

__device__ inline void async_copy_b128(const void* gsrc, void* lds_dst) {
#if defined(__has_builtin) && __has_builtin(__builtin_amdgcn_global_load_async_to_lds_b128)
  __builtin_amdgcn_global_load_async_to_lds_b128(
      (as1_v4i_ptr)gsrc, (as3_v4i_ptr)lds_dst, 0, 0);
#else
  *(uint4*)lds_dst = *(const uint4*)gsrc;  // plain load + ds_store fallback
#endif
}

__device__ inline void wait_async0() {
#if defined(__has_builtin) && __has_builtin(__builtin_amdgcn_s_wait_asynccnt)
  __builtin_amdgcn_s_wait_asynccnt(0);
#else
  asm volatile("s_wait_asynccnt 0x0" ::: "memory");
#endif
}

// f32 -> f16 convert with zero row padding (also used for x itself).
__global__ void pack_f16_kernel(const float* __restrict__ src,
                                _Float16* __restrict__ dst,
                                int Nvalid, int K, int total) {
  int i = blockIdx.x * blockDim.x + threadIdx.x;
  if (i >= total) return;
  int n = i / K;
  int k = i - n * K;
  float v = (n < Nvalid) ? src[(size_t)n * K + k] : 0.0f;
  dst[i] = (_Float16)v;
}

// Row-normalized Gaussian RBF basis, stored padded [F][16].
__global__ void basis_kernel(float* __restrict__ basis) {
  const int f = blockIdx.x * blockDim.x + threadIdx.x;
  if (f >= 1024) return;
  const float t = (float)f * (1.0f / 1023.0f);
  float v[10];
  float s = 0.0f;
#pragma unroll
  for (int p = 0; p < 10; ++p) {
    float d = t - (float)p * (1.0f / 9.0f);
    v[p] = expf(-d * d * 50.0f);   // sigma = 1/10 -> 1/(2 sigma^2) = 50
    s += v[p];
  }
  const float inv = 1.0f / (s + 1e-8f);
#pragma unroll
  for (int p = 0; p < 10; ++p) basis[f * 16 + p] = v[p] * inv;
#pragma unroll
  for (int p = 10; p < 16; ++p) basis[f * 16 + p] = 0.0f;
}

// C = act(A[M,K] @ W[N,K]^T + bias), K = KT*32, N = NT*16 (compile-time).
// Block: 128 threads = 4 waves; wave w owns M strip [blk*64 + w*16, +16).
// W async-staged to LDS; bias zero-padded into LDS (branchless epilogue).
template <int KT, int NT, int ACT, bool OUT16>
__global__ __launch_bounds__(128) void gemm_ker(const _Float16* __restrict__ A,
                                                const _Float16* __restrict__ W,
                                                const float* __restrict__ bias,
                                                void* __restrict__ outv,
                                                int Nvalid) {
  constexpr int K = KT * 32;
  constexpr int N = NT * 16;
  extern __shared__ char smem_raw[];
  _Float16* sW = (_Float16*)smem_raw;
  float* sBias = (float*)(smem_raw + N * K * 2);

  // Cooperative async stage of W ([N][K] f16, N*K*2 bytes) into LDS.
  constexpr int CHUNKS = (N * K * 2) / 16;
  for (int c = threadIdx.x; c < CHUNKS; c += 128)
    async_copy_b128((const char*)W + (size_t)c * 16, smem_raw + (size_t)c * 16);
  // Zero-padded bias into LDS (plain ds stores, DScnt).
  for (int n = threadIdx.x; n < N; n += 128)
    sBias[n] = (n < Nvalid) ? bias[n] : 0.0f;
  wait_async0();
  __syncthreads();

  const int lane = threadIdx.x & 31;
  const int wave = threadIdx.x >> 5;
  const int hi = lane >> 4;
  const int lr = lane & 15;
  const int mrow = blockIdx.x * 64 + wave * 16;

  // A fragments for this wave's 16-row strip: registers for the whole N sweep.
  v16h af[KT];
  const _Float16* arow = A + (size_t)(mrow + lr) * K;
#pragma unroll
  for (int kt = 0; kt < KT; ++kt) af[kt] = load_frag16(arow + kt * 32, hi);

  for (int nt = 0; nt < NT; ++nt) {
    const int n0 = nt * 16;
    const int ncol = n0 + lr;
    const _Float16* swrow = sW + (size_t)ncol * K;
    v8f c = {};
#pragma unroll
    for (int kt = 0; kt < KT; ++kt) {
      v16h bf = load_frag16(swrow + kt * 32, hi);   // ds_load_b128 x2
      c = __builtin_amdgcn_wmma_f32_16x16x32_f16(
          false, af[kt], false, bf, (short)0, c, false, false);
    }
    const float bv = sBias[ncol];                   // ds_load_b32, branchless
    // C layout: lane holds col n = lr; vgpr r holds row m = r + 8*hi.
    if (OUT16) {
      _Float16* ob = (_Float16*)outv + (size_t)(mrow + 8 * hi) * N + ncol;
#pragma unroll
      for (int r = 0; r < 8; ++r) {
        float v = c[r] + bv;
        if (ACT == 1) v = v > 0.0f ? v : 0.0f;
        ob[(size_t)r * N] = (_Float16)v;   // immediate offsets (N constexpr)
      }
    } else {
      float* ob = (float*)outv + (size_t)(mrow + 8 * hi) * N + ncol;
#pragma unroll
      for (int r = 0; r < 8; ++r) {
        float v = c[r] + bv;
        if (ACT == 1) v = v > 0.0f ? v : 0.0f;
        ob[(size_t)r * N] = v;
      }
    }
  }
}

// Per-batch-row: tanh + control-point mean, softmax. f32 math.
__global__ void finalize_kernel(const float* __restrict__ cpraw,  // [B][32] (20 valid)
                                const float* __restrict__ wraw,   // [B][16] (10 valid)
                                float* __restrict__ wa,           // [B][16] w*cpm
                                float* __restrict__ ww,           // [B][16] w
                                int B) {
  int b = blockIdx.x * blockDim.x + threadIdx.x;
  if (b >= B) return;
  float w[10];
  float mx = -1e30f;
#pragma unroll
  for (int p = 0; p < 10; ++p) {
    w[p] = wraw[(size_t)b * 16 + p];
    mx = fmaxf(mx, w[p]);
  }
  float s = 0.0f;
#pragma unroll
  for (int p = 0; p < 10; ++p) {
    w[p] = expf(w[p] - mx);
    s += w[p];
  }
  const float inv = 1.0f / s;
#pragma unroll
  for (int p = 0; p < 10; ++p) {
    float c0 = tanhf(cpraw[(size_t)b * 32 + 2 * p]);
    float c1 = tanhf(cpraw[(size_t)b * 32 + 2 * p + 1]);
    float cm = 0.5f * (c0 + c1);
    float wp = w[p] * inv;
    ww[(size_t)b * 16 + p] = wp;
    wa[(size_t)b * 16 + p] = wp * cm;
  }
}

// out[b,f] = (basis[f,:].wa[b,:]) / (basis[f,:].ww[b,:] + eps)
// 256 threads own 256 consecutive f (basis rows in registers); b is
// block-uniform inside the loop -> wa/ww loads become scalar broadcasts.
__global__ __launch_bounds__(256) void combine_kernel(
    const float* __restrict__ basis, const float* __restrict__ wa,
    const float* __restrict__ ww, float* __restrict__ out, int F) {
  const int f = blockIdx.x * 256 + threadIdx.x;
  float r[10];
#pragma unroll
  for (int p = 0; p < 10; ++p) r[p] = basis[(size_t)f * 16 + p];
  const int b0 = blockIdx.y * 16;
#pragma unroll 4
  for (int bb = 0; bb < 16; ++bb) {
    const int b = b0 + bb;
    const float* pwa = wa + (size_t)b * 16;
    const float* pww = ww + (size_t)b * 16;
    float num = 0.0f, den = 1e-8f;
#pragma unroll
    for (int p = 0; p < 10; ++p) {
      num += r[p] * pwa[p];
      den += r[p] * pww[p];
    }
    out[(size_t)b * F + f] = num / den;
  }
}

extern "C" void kernel_launch(void* const* d_in, const int* in_sizes, int n_in,
                              void* d_out, int out_size, void* d_ws, size_t ws_size,
                              hipStream_t stream) {
  (void)in_sizes; (void)n_in; (void)out_size; (void)ws_size;
  const int B = 16384, D = 128, F = 1024;

  const float* x     = (const float*)d_in[0];
  const float* cp_w1 = (const float*)d_in[1];
  const float* cp_b1 = (const float*)d_in[2];
  const float* cp_w2 = (const float*)d_in[3];
  const float* cp_b2 = (const float*)d_in[4];
  const float* cp_w3 = (const float*)d_in[5];
  const float* cp_b3 = (const float*)d_in[6];
  const float* w_w1  = (const float*)d_in[7];
  const float* w_b1  = (const float*)d_in[8];
  const float* w_w2  = (const float*)d_in[9];
  const float* w_b2  = (const float*)d_in[10];
  const float* w_w3  = (const float*)d_in[11];
  const float* w_b3  = (const float*)d_in[12];
  float* out = (float*)d_out;

  char* ws = (char*)d_ws;
  size_t off = 0;
  auto take = [&](size_t bytes) -> char* {
    char* p = ws + off;
    off = (off + bytes + 255) & ~(size_t)255;
    return p;
  };
  _Float16* x16   = (_Float16*)take((size_t)B * D * 2);
  _Float16* cw1   = (_Float16*)take((size_t)128 * 128 * 2);
  _Float16* cw2   = (_Float16*)take((size_t)256 * 128 * 2);
  _Float16* cw3   = (_Float16*)take((size_t)32 * 256 * 2);   // 20 -> pad 32
  _Float16* vw1   = (_Float16*)take((size_t)64 * 128 * 2);
  _Float16* vw2   = (_Float16*)take((size_t)128 * 64 * 2);
  _Float16* vw3   = (_Float16*)take((size_t)16 * 128 * 2);   // 10 -> pad 16
  _Float16* h1c   = (_Float16*)take((size_t)B * 128 * 2);
  _Float16* h2c   = (_Float16*)take((size_t)B * 256 * 2);
  float*    cpraw = (float*)take((size_t)B * 32 * 4);
  _Float16* h1w   = (_Float16*)take((size_t)B * 64 * 2);
  _Float16* h2w   = (_Float16*)take((size_t)B * 128 * 2);
  float*    wraw  = (float*)take((size_t)B * 16 * 4);
  float*    wa    = (float*)take((size_t)B * 16 * 4);
  float*    wwp   = (float*)take((size_t)B * 16 * 4);
  float*    basis = (float*)take((size_t)F * 16 * 4);

  auto pk = [&](const float* s, _Float16* d, int Nv, int K, int Npad) {
    int tot = Npad * K;
    pack_f16_kernel<<<(tot + 255) / 256, 256, 0, stream>>>(s, d, Nv, K, tot);
  };
  pk(x, x16, B, D, B);
  pk(cp_w1, cw1, 128, 128, 128);
  pk(cp_w2, cw2, 256, 128, 256);
  pk(cp_w3, cw3, 20, 256, 32);
  pk(w_w1, vw1, 64, 128, 64);
  pk(w_w2, vw2, 128, 64, 128);
  pk(w_w3, vw3, 10, 128, 16);
  basis_kernel<<<4, 256, 0, stream>>>(basis);

  const dim3 blk(128, 1, 1);
  const dim3 grd(B / 64, 1, 1);
  // shmem = N*K*2 (weights) + N*4 (padded bias) bytes per layer
  gemm_ker<4,  8, 1, true ><<<grd, blk, 128 * 128 * 2 + 128 * 4, stream>>>(x16, cw1, cp_b1, h1c, 128);
  gemm_ker<4, 16, 1, true ><<<grd, blk, 256 * 128 * 2 + 256 * 4, stream>>>(h1c, cw2, cp_b2, h2c, 256);
  gemm_ker<8,  2, 0, false><<<grd, blk,  32 * 256 * 2 +  32 * 4, stream>>>(h2c, cw3, cp_b3, cpraw, 20);
  gemm_ker<4,  4, 1, true ><<<grd, blk,  64 * 128 * 2 +  64 * 4, stream>>>(x16, vw1, w_b1, h1w, 64);
  gemm_ker<2,  8, 1, true ><<<grd, blk, 128 *  64 * 2 + 128 * 4, stream>>>(h1w, vw2, w_b2, h2w, 128);
  gemm_ker<4,  1, 0, false><<<grd, blk,  16 * 128 * 2 +  16 * 4, stream>>>(h2w, vw3, w_b3, wraw, 10);

  finalize_kernel<<<B / 256, 256, 0, stream>>>(cpraw, wraw, wa, wwp, B);

  combine_kernel<<<dim3(F / 256, B / 16), 256, 0, stream>>>(basis, wa, wwp, out, F);
}
